// TreeLSTM_4526895530471
// MI455X (gfx1250) — matrix operand order repaired and checked by reading, not compiled, not run on hardware
//
#include <hip/hip_runtime.h>
#include <hip/hip_bf16.h>

#define B_TOT 4096
#define NCH   32
#define ZSIZE 512
#define CH    256
#define KTOT  768   // 512 (z/W part) + 256 (h/U part)
#define KT    32    // K per WMMA step
#define LDSTR 40    // LDS row stride in halves (32 + 8 pad -> conflict-free)

typedef __bf16 v16bf __attribute__((ext_vector_type(16)));
typedef float  v8f   __attribute__((ext_vector_type(8)));
typedef unsigned short v8u16  __attribute__((ext_vector_type(8)));
typedef unsigned short v16u16 __attribute__((ext_vector_type(16)));

__device__ __forceinline__ unsigned short f2bf(float f) {
    unsigned u = __builtin_bit_cast(unsigned, f);
    u += 0x7FFFu + ((u >> 16) & 1u);          // round-to-nearest-even
    return (unsigned short)(u >> 16);
}

// load 8 consecutive f32 (two b128 loads), convert to 8 bf16
__device__ __forceinline__ v8u16 ldcvt8(const float* __restrict__ src) {
    float4 f0 = *(const float4*)src;
    float4 f1 = *(const float4*)(src + 4);
    v8u16 r;
    r[0] = f2bf(f0.x); r[1] = f2bf(f0.y); r[2] = f2bf(f0.z); r[3] = f2bf(f0.w);
    r[4] = f2bf(f1.x); r[5] = f2bf(f1.y); r[6] = f2bf(f1.z); r[7] = f2bf(f1.w);
    return r;
}

// A fragment (16x32, row per lane): lane<16 -> K {0..7,16..23}; lane>=16 -> K {8..15,24..31}
__device__ __forceinline__ v16bf frag_ld_a(const unsigned short* __restrict__ rowbase, int half) {
    const int koff = half * 8;
    v8u16 lo = *(const v8u16*)(rowbase + koff);
    v8u16 hi = *(const v8u16*)(rowbase + koff + 16);
    v16u16 c = __builtin_shufflevector(lo, hi, 0,1,2,3,4,5,6,7,8,9,10,11,12,13,14,15);
    return __builtin_bit_cast(v16bf, c);
}

// B fragment (32x16, column per lane): lane<16 -> K {0..7,16..23}; lane>=16 -> K {8..15,24..31}
__device__ __forceinline__ v16bf frag_ld_b(const unsigned short* __restrict__ rowbase, int half) {
    const int koff = half * 8;
    v8u16 lo = *(const v8u16*)(rowbase + koff);
    v8u16 hi = *(const v8u16*)(rowbase + koff + 16);
    v16u16 c = __builtin_shufflevector(lo, hi, 0,1,2,3,4,5,6,7,8,9,10,11,12,13,14,15);
    return __builtin_bit_cast(v16bf, c);
}

// fast transcendentals on v_exp_f32 / v_rcp_f32
__device__ __forceinline__ float fast_sigmoid(float x) {
    float e = __builtin_amdgcn_exp2f(x * -1.44269504f);
    return __builtin_amdgcn_rcpf(1.0f + e);
}
__device__ __forceinline__ float fast_tanh(float x) {
    float xc = fminf(fmaxf(x, -15.0f), 15.0f);
    float e  = __builtin_amdgcn_exp2f(xc * 2.88539008f);   // exp(2x)
    return (e - 1.0f) * __builtin_amdgcn_rcpf(e + 1.0f);
}

// -------- kernel 1: h_tilde[b,c] = sum_n z_children[b,n,c]  (c < CH) --------
__global__ void __launch_bounds__(256)
k_htilde(const float* __restrict__ z_ch, float* __restrict__ htilde) {
    const int b = blockIdx.x;
    const int c = threadIdx.x;                  // 0..255
    const float* p = z_ch + (size_t)b * NCH * ZSIZE + c;
    float s = 0.0f;
    #pragma unroll
    for (int n = 0; n < NCH; ++n) s += p[(size_t)n * ZSIZE];
    htilde[(size_t)b * CH + c] = s;
}

// tile source helpers ---------------------------------------------------------
__device__ __forceinline__ const float* a_src_gates(const float* __restrict__ z_node,
                                                    const float* __restrict__ htilde,
                                                    int b, int kcol) {
    return (kcol < ZSIZE) ? (z_node + (size_t)b * NCH * ZSIZE + kcol)   // z0 = z_node[:,0]
                          : (htilde + (size_t)b * CH + (kcol - ZSIZE));
}
__device__ __forceinline__ const float* a_src_f(const float* __restrict__ z_node,
                                                const float* __restrict__ z_ch,
                                                int gr, int kcol) {
    return (kcol < ZSIZE) ? (z_node + (size_t)gr * ZSIZE + kcol)
                          : (z_ch   + (size_t)gr * ZSIZE + (kcol - ZSIZE)); // h_k
}
__device__ __forceinline__ const float* b_src(const float* __restrict__ W,
                                              const float* __restrict__ U,
                                              int c, int kcol) {
    return (kcol < ZSIZE) ? (W + (size_t)c * ZSIZE + kcol)
                          : (U + (size_t)c * CH + (kcol - ZSIZE));
}

// -------- kernel 2: gates i/o/u: act([z0|h_tilde] @ [W;U]^T + bias) ----------
// grid = (B/64, 3 gates), block = 256 (8 waves: 2 in M x 4 in N)
__global__ void __launch_bounds__(256)
k_gates(const float* __restrict__ z_node, const float* __restrict__ htilde,
        const float* __restrict__ Wi, const float* __restrict__ Ui, const float* __restrict__ bi,
        const float* __restrict__ Wo, const float* __restrict__ Uo, const float* __restrict__ bo,
        const float* __restrict__ Wu, const float* __restrict__ Uu, const float* __restrict__ bu,
        float* __restrict__ ws_act)
{
    __shared__ __align__(16) unsigned short sA[64  * LDSTR];
    __shared__ __align__(16) unsigned short sB[256 * LDSTR];

    const int t     = threadIdx.x;
    const int bbase = blockIdx.x * 64;
    const int g     = blockIdx.y;
    const float* W  = (g == 0) ? Wi : (g == 1) ? Wo : Wu;
    const float* U  = (g == 0) ? Ui : (g == 1) ? Uo : Uu;
    const float* bg = (g == 0) ? bi : (g == 1) ? bo : bu;

    const int lane  = t & 31, wave = t >> 5;
    const int waveM = wave >> 2, waveN = wave & 3;   // 2 x 4 wave grid
    const int half  = lane >> 4, lr = lane & 15;

    const int m_ld  = t >> 2;           // A row / B row-base this thread stages
    const int kk_ld = (t & 3) * 8;

    v8f acc[2][4] = {};

    // prologue: stage k-tile 0 into registers
    v8u16 ra = ldcvt8(a_src_gates(z_node, htilde, bbase + m_ld, kk_ld));
    v8u16 rb[4];
    #pragma unroll
    for (int i = 0; i < 4; ++i)
        rb[i] = ldcvt8(b_src(W, U, m_ld + 64 * i, kk_ld));

    for (int kt = 0; kt < KTOT; kt += KT) {
        // commit staged registers to LDS
        *(v8u16*)&sA[m_ld * LDSTR + kk_ld] = ra;
        #pragma unroll
        for (int i = 0; i < 4; ++i)
            *(v8u16*)&sB[(m_ld + 64 * i) * LDSTR + kk_ld] = rb[i];
        __syncthreads();

        // prefetch next k-tile (uniform branch, EXEC stays all-ones)
        const int ktn = kt + KT;
        if (ktn < KTOT) {
            ra = ldcvt8(a_src_gates(z_node, htilde, bbase + m_ld, ktn + kk_ld));
            #pragma unroll
            for (int i = 0; i < 4; ++i)
                rb[i] = ldcvt8(b_src(W, U, m_ld + 64 * i, ktn + kk_ld));
        }

        v16bf af[2], bf2[4];
        #pragma unroll
        for (int mt = 0; mt < 2; ++mt)
            af[mt] = frag_ld_a(&sA[(waveM * 32 + mt * 16 + lr) * LDSTR], half);
        #pragma unroll
        for (int nt = 0; nt < 4; ++nt)
            bf2[nt] = frag_ld_b(&sB[(waveN * 64 + nt * 16 + lr) * LDSTR], half);
        #pragma unroll
        for (int mt = 0; mt < 2; ++mt)
            #pragma unroll
            for (int nt = 0; nt < 4; ++nt)
                acc[mt][nt] = __builtin_amdgcn_wmma_f32_16x16x32_bf16(
                    false, af[mt], false, bf2[nt], (short)0, acc[mt][nt], false, false);
        __syncthreads();
    }

    // epilogue: activation -> ws_act[g][b][c]
    float* dst = ws_act + (size_t)g * B_TOT * CH;
    #pragma unroll
    for (int nt = 0; nt < 4; ++nt) {
        int c = waveN * 64 + nt * 16 + lr;
        float bias = bg[c];
        #pragma unroll
        for (int mt = 0; mt < 2; ++mt) {
            #pragma unroll
            for (int j = 0; j < 8; ++j) {
                int b = bbase + waveM * 32 + mt * 16 + j + half * 8;
                float v = acc[mt][nt][j] + bias;
                float a = (g == 2) ? fast_tanh(v) : fast_sigmoid(v);
                dst[(size_t)b * CH + c] = a;
            }
        }
    }
}

// -------- kernel 3: f-gate GEMM + fused tree reduction + output --------------
// rows g = tree*32 + n; M-tile = 64 rows (2 trees), N-tile = 256 (all channels)
// grid = B/2, block = 256 (8 waves: waveM picks the tree, waveN the 64-col slab)
__global__ void __launch_bounds__(256)
k_fgate(const float* __restrict__ z_node, const float* __restrict__ z_ch,
        const float* __restrict__ Wf, const float* __restrict__ Uf,
        const float* __restrict__ bf_, const float* __restrict__ ws_act,
        float* __restrict__ out)
{
    __shared__ __align__(16) unsigned short sA[64  * LDSTR];
    __shared__ __align__(16) unsigned short sB[256 * LDSTR];

    const int t     = threadIdx.x;
    const int rbase = blockIdx.x * 64;
    const int lane  = t & 31, wave = t >> 5;
    const int waveM = wave >> 2, waveN = wave & 3;
    const int half  = lane >> 4, lr = lane & 15;

    const int m_ld  = t >> 2;
    const int kk_ld = (t & 3) * 8;

    v8f acc[2][4] = {};

    v8u16 ra = ldcvt8(a_src_f(z_node, z_ch, rbase + m_ld, kk_ld));
    v8u16 rb[4];
    #pragma unroll
    for (int i = 0; i < 4; ++i)
        rb[i] = ldcvt8(b_src(Wf, Uf, m_ld + 64 * i, kk_ld));

    for (int kt = 0; kt < KTOT; kt += KT) {
        *(v8u16*)&sA[m_ld * LDSTR + kk_ld] = ra;
        #pragma unroll
        for (int i = 0; i < 4; ++i)
            *(v8u16*)&sB[(m_ld + 64 * i) * LDSTR + kk_ld] = rb[i];
        __syncthreads();

        const int ktn = kt + KT;
        if (ktn < KTOT) {
            ra = ldcvt8(a_src_f(z_node, z_ch, rbase + m_ld, ktn + kk_ld));
            #pragma unroll
            for (int i = 0; i < 4; ++i)
                rb[i] = ldcvt8(b_src(Wf, Uf, m_ld + 64 * i, ktn + kk_ld));
        }

        v16bf af[2], bf2[4];
        #pragma unroll
        for (int mt = 0; mt < 2; ++mt)
            af[mt] = frag_ld_a(&sA[(waveM * 32 + mt * 16 + lr) * LDSTR], half);
        #pragma unroll
        for (int nt = 0; nt < 4; ++nt)
            bf2[nt] = frag_ld_b(&sB[(waveN * 64 + nt * 16 + lr) * LDSTR], half);
        #pragma unroll
        for (int mt = 0; mt < 2; ++mt)
            #pragma unroll
            for (int nt = 0; nt < 4; ++nt)
                acc[mt][nt] = __builtin_amdgcn_wmma_f32_16x16x32_bf16(
                    false, af[mt], false, bf2[nt], (short)0, acc[mt][nt], false, false);
        __syncthreads();
    }

    // epilogue: per wave, one tree; sum over its 32 children of sigmoid(f)*c_k,
    // then fuse c_j = i*u + sum, h_j = o*tanh(c_j)
    const int tree = blockIdx.x * 2 + waveM;
    const float* ai = ws_act + (size_t)0 * B_TOT * CH;
    const float* ao = ws_act + (size_t)1 * B_TOT * CH;
    const float* au = ws_act + (size_t)2 * B_TOT * CH;

    #pragma unroll
    for (int nt = 0; nt < 4; ++nt) {
        int col  = waveN * 64 + nt * 16 + lr;
        float bias = bf_[col];
        float s = 0.0f;
        #pragma unroll
        for (int mt = 0; mt < 2; ++mt) {
            #pragma unroll
            for (int j = 0; j < 8; ++j) {
                int n = mt * 16 + j + half * 8;       // child index this lane holds
                float ck = z_ch[(size_t)(tree * NCH + n) * ZSIZE + CH + col];
                float fg = fast_sigmoid(acc[mt][nt][j] + bias);
                s += fg * ck;
            }
        }
        s += __shfl_xor(s, 16, 32);                   // combine complementary rows
        if (half == 0) {
            size_t idx = (size_t)tree * CH + col;
            float cj = ai[idx] * au[idx] + s;
            float hj = ao[idx] * fast_tanh(cj);
            out[(size_t)tree * ZSIZE + col]      = hj;
            out[(size_t)tree * ZSIZE + CH + col] = cj;
        }
    }
}

extern "C" void kernel_launch(void* const* d_in, const int* in_sizes, int n_in,
                              void* d_out, int out_size, void* d_ws, size_t ws_size,
                              hipStream_t stream) {
    (void)in_sizes; (void)n_in; (void)out_size; (void)ws_size;
    const float* z_node = (const float*)d_in[0];
    const float* z_ch   = (const float*)d_in[1];
    const float* Wi = (const float*)d_in[2],  *Ui = (const float*)d_in[3],  *bi = (const float*)d_in[4];
    const float* Wf = (const float*)d_in[5],  *Uf = (const float*)d_in[6],  *bf = (const float*)d_in[7];
    const float* Wo = (const float*)d_in[8],  *Uo = (const float*)d_in[9],  *bo = (const float*)d_in[10];
    const float* Wu = (const float*)d_in[11], *Uu = (const float*)d_in[12], *bu = (const float*)d_in[13];

    float* out    = (float*)d_out;
    float* htilde = out + (size_t)B_TOT * ZSIZE;   // second tuple output lives here
    float* ws_act = (float*)d_ws;                  // 3 * B*CH f32 = 12 MB scratch

    k_htilde<<<B_TOT, 256, 0, stream>>>(z_ch, htilde);
    k_gates<<<dim3(B_TOT / 64, 3), 256, 0, stream>>>(z_node, htilde,
                                                     Wi, Ui, bi, Wo, Uo, bo, Wu, Uu, bu,
                                                     ws_act);
    k_fgate<<<B_TOT / 2, 256, 0, stream>>>(z_node, z_ch, Wf, Uf, bf, ws_act, out);
}
